// CausalSelfAttention_36584531427532
// MI455X (gfx1250) — matrix operand order repaired and checked by reading, not compiled, not run on hardware
//
#include <hip/hip_runtime.h>
#include <hip/hip_bf16.h>

// ---------- CDNA5 WMMA types ----------
typedef __attribute__((ext_vector_type(16))) __bf16       v16bf;
typedef __attribute__((ext_vector_type(8)))  float        v8f;
typedef __attribute__((ext_vector_type(4)))  unsigned int uint4v;
typedef __attribute__((ext_vector_type(4)))  int          int4v;

typedef __attribute__((address_space(1))) int4v g_int4;   // global
typedef __attribute__((address_space(3))) int4v l_int4;   // LDS

// f32 -> bf16 bits (round-to-nearest-even-ish)
__device__ __forceinline__ unsigned short f2bf(float x) {
    unsigned int u = __float_as_uint(x);
    unsigned int r = u + 0x7FFFu + ((u >> 16) & 1u);
    return (unsigned short)(r >> 16);
}

// 16-byte global -> LDS copy via ASYNCcnt (no VGPR data path).
__device__ __forceinline__ void async_copy16(const unsigned short* g, unsigned short* l) {
#if __has_builtin(__builtin_amdgcn_global_load_async_to_lds_b128)
    __builtin_amdgcn_global_load_async_to_lds_b128((g_int4*)g, (l_int4*)l, 0, 0);
#else
    *(uint4v*)l = *(const uint4v*)g;
#endif
}

__device__ __forceinline__ void wait_async() {
#if __has_builtin(__builtin_amdgcn_s_wait_asynccnt)
    __builtin_amdgcn_s_wait_asynccnt(0);
#else
    asm volatile("s_wait_asynccnt 0x0" ::: "memory");
#endif
}

// A-matrix fragment (16x32 bf16, MxK): lane = row; lo/hi lane half selects K 0-7/16-23 vs 8-15/24-31
__device__ __forceinline__ v16bf load_a_frag(const unsigned short* lds, int row, int stride, int kbase) {
    union { v16bf v; uint4v q[2]; } f;
    f.q[0] = *(const uint4v*)(lds + row * stride + kbase);
    f.q[1] = *(const uint4v*)(lds + row * stride + kbase + 16);
    return f.v;
}

// B-matrix fragment (32x16 bf16, KxN) from a B^T ([n][k]) LDS layout
__device__ __forceinline__ v16bf load_b_frag(const unsigned short* lds, int col, int stride, int kstart) {
    union { v16bf v; uint4v q[2]; } f;
    f.q[0] = *(const uint4v*)(lds + col * stride + kstart);
    f.q[1] = *(const uint4v*)(lds + col * stride + kstart + 16);
    return f.v;
}

__device__ __forceinline__ v8f wmma_bf16(v16bf a, v16bf b, v8f c) {
    return __builtin_amdgcn_wmma_f32_16x16x32_bf16(false, a, false, b, (short)0, c, false, false);
}

// =====================================================================
// Prep kernels: all f32->bf16 conversion / transposition hoisted here.
// =====================================================================
__global__ __launch_bounds__(256) void cvt_bf16_kernel(
    const float* __restrict__ src, unsigned short* __restrict__ dst, int n)
{
    int i = blockIdx.x * 256 + threadIdx.x;
    if (i < n) dst[i] = f2bf(src[i]);
}

// dst[n*K + k] = bf16(src[k*N + n])
__global__ __launch_bounds__(256) void transpose_cvt_kernel(
    const float* __restrict__ src, unsigned short* __restrict__ dst, int K, int N)
{
    __shared__ float tile[32][33];
    const int k0 = blockIdx.y * 32, n0 = blockIdx.x * 32;
    const int tx = threadIdx.x & 31, ty = threadIdx.x >> 5;
    for (int yy = ty; yy < 32; yy += 8)
        tile[yy][tx] = src[(size_t)(k0 + yy) * N + n0 + tx];
    __syncthreads();
    for (int yy = ty; yy < 32; yy += 8)
        dst[(size_t)(n0 + yy) * K + k0 + tx] = f2bf(tile[tx][yy]);
}

// V (B,T,4,128) f32 -> Vt (B,4,128,T) bf16
__global__ __launch_bounds__(256) void vtrans_kernel(
    const float* __restrict__ V, unsigned short* __restrict__ Vt)
{
    __shared__ float tile[32][33];
    const int s0 = blockIdx.x * 32, d0 = blockIdx.y * 32;
    const int bk = blockIdx.z;
    const int b  = bk >> 2, kv = bk & 3;
    const int tx = threadIdx.x & 31, ty = threadIdx.x >> 5;
    for (int yy = ty; yy < 32; yy += 8)
        tile[yy][tx] = V[((size_t)(b * 2048 + s0 + yy) * 4 + kv) * 128 + d0 + tx];
    __syncthreads();
    for (int yy = ty; yy < 32; yy += 8)
        Vt[((size_t)bk * 128 + d0 + yy) * 2048 + s0 + tx] = f2bf(tile[tx][yy]);
}

// =====================================================================
// GEMM: C[MxN] f32 = A[MxK] bf16 @ Bt[NxK] bf16. Block tile 128x128,
// K-step 32, double-buffered LDS: async copies of tile k+1 overlap the
// 8 WMMAs of tile k. N/Kd are template constants -> immediate-offset
// addressing in staging and epilogue.
// =====================================================================
template<int N, int Kd>
__global__ __launch_bounds__(256) void gemm_bf16_kernel(
    const unsigned short* __restrict__ A, const unsigned short* __restrict__ Bt,
    float* __restrict__ C)
{
    __shared__ unsigned short shA[2][128 * 32];
    __shared__ unsigned short shB[2][128 * 32];

    const int tid  = threadIdx.x;
    const int w    = tid >> 5;
    const int lane = tid & 31;
    const int lh   = lane & 15;
    const bool hi  = lane >= 16;
    const int wr   = w >> 1;                    // 0..3 : 32-row strip
    const int wc   = w & 1;                     // 0..1 : 64-col strip
    const int row0 = blockIdx.y * 128;
    const int col0 = blockIdx.x * 128;

    // per-thread staging bases: 2 x 16B chunks each for A and B per tile
    const int sr = tid >> 2;                    // 0..63
    const int sc = (tid & 3) * 8;               // 0,8,16,24
    const unsigned short* gA = A  + (size_t)(row0 + sr) * Kd + sc;
    const unsigned short* gB = Bt + (size_t)(col0 + sr) * Kd + sc;
    const int lofs = sr * 32 + sc;

    v8f acc[2][4] = {};

    auto issue_tiles = [&](int k0, int buf) {
        async_copy16(gA + k0,                 &shA[buf][lofs]);
        async_copy16(gA + (size_t)64 * Kd + k0, &shA[buf][lofs + 64 * 32]);
        async_copy16(gB + k0,                 &shB[buf][lofs]);
        async_copy16(gB + (size_t)64 * Kd + k0, &shB[buf][lofs + 64 * 32]);
    };

    issue_tiles(0, 0);
    wait_async();
    __syncthreads();

    constexpr int NK = Kd / 32;
    for (int kk = 0; kk < NK; ++kk) {
        const int cur = kk & 1;
        if (kk + 1 < NK) issue_tiles((kk + 1) * 32, cur ^ 1);   // overlap with WMMA

        const unsigned short* pA = shA[cur];
        const unsigned short* pB = shB[cur];
        v16bf a0 = load_a_frag(pA, wr * 32 + lh,      32, hi ? 8 : 0);
        v16bf a1 = load_a_frag(pA, wr * 32 + 16 + lh, 32, hi ? 8 : 0);
        #pragma unroll
        for (int st = 0; st < 4; ++st) {
            v16bf bfr = load_b_frag(pB, wc * 64 + st * 16 + lh, 32, hi ? 8 : 0);
            acc[0][st] = wmma_bf16(a0, bfr, acc[0][st]);
            acc[1][st] = wmma_bf16(a1, bfr, acc[1][st]);
        }

        wait_async();
        __syncthreads();
    }

    #pragma unroll
    for (int rr = 0; rr < 2; ++rr)
        #pragma unroll
        for (int st = 0; st < 4; ++st)
            #pragma unroll
            for (int e = 0; e < 8; ++e)
                C[(size_t)(row0 + wr * 32 + rr * 16 + (hi ? 8 : 0) + e) * N +
                  col0 + wc * 64 + st * 16 + lh] = acc[rr][st][e];
}

// =====================================================================
// RoPE + RMS-norm: f32 in, bf16 out (1/sqrt(d) folded into Q's scale).
// =====================================================================
__global__ __launch_bounds__(64) void rope_rms_kernel(
    const float* __restrict__ in, unsigned short* __restrict__ outb,
    const float* __restrict__ cosb, const float* __restrict__ sinb, int nh, float scale)
{
    __shared__ float red[64];
    const int bid = blockIdx.x;
    const int h = bid % nh;
    const int t = (bid / nh) % 2048;
    const int b = bid / (nh * 2048);
    const size_t base = ((size_t)(b * 2048 + t) * nh + h) * 128;

    const int i = threadIdx.x;
    const float c = cosb[t * 64 + i];
    const float s = sinb[t * 64 + i];
    const float x1 = in[base + i];
    const float x2 = in[base + 64 + i];
    const float o1 =  x1 * c + x2 * s;
    const float o2 = -x1 * s + x2 * c;

    red[i] = o1 * o1 + o2 * o2;
    __syncthreads();
    for (int off = 32; off > 0; off >>= 1) {
        if (i < off) red[i] += red[i + off];
        __syncthreads();
    }
    const float inv = rsqrtf(red[0] * (1.0f / 128.0f) + 1.1920928955078125e-07f) * scale;
    outb[base + i]      = f2bf(o1 * inv);
    outb[base + 64 + i] = f2bf(o2 * inv);
}

// =====================================================================
// Flash-attention: 4 waves x 16 Q rows, KV steps of 32, GQA (16Q/4KV).
// Double-buffered K/V tiles: async copy of step s+1 overlaps the
// QK^T WMMA + softmax + PV WMMA of step s.
// =====================================================================
__global__ __launch_bounds__(128) void attn_kernel(
    const unsigned short* __restrict__ Qb, const unsigned short* __restrict__ Kb,
    const unsigned short* __restrict__ Vt, unsigned short* __restrict__ Yb)
{
    __shared__ unsigned short shQ[64 * 128];       // A layout [r][d]
    __shared__ unsigned short shK[2][32 * 128];    // [s][d]
    __shared__ unsigned short shVT[2][128 * 32];   // [d][s]
    __shared__ float          shS[64 * 32];
    __shared__ unsigned short shP[64 * 32];
    __shared__ float          shCorr[64];
    __shared__ float          shL[64];

    const int tid  = threadIdx.x;
    const int w    = tid >> 5;
    const int lane = tid & 31;
    const int lh   = lane & 15;
    const bool hi  = lane >= 16;

    const int m0 = blockIdx.x * 64;
    const int h  = blockIdx.y;
    const int b  = blockIdx.z;
    const int kv = h >> 2;

    // staging bases
    const unsigned short* gQ = Qb + ((size_t)(b * 2048 + m0 + (tid >> 4)) * 16 + h) * 128 + (tid & 15) * 8;
    const unsigned short* gK = Kb + ((size_t)(b * 2048 + (tid >> 4)) * 4 + kv) * 128 + (tid & 15) * 8;
    const unsigned short* gV = Vt + ((size_t)(b * 4 + kv) * 128 + (tid >> 2)) * 2048 + (tid & 3) * 8;
    const int lK = (tid >> 4) * 128 + (tid & 15) * 8;
    const int lV = (tid >> 2) * 32 + (tid & 3) * 8;

    auto issue_kv = [&](int s0, int buf) {
        #pragma unroll
        for (int j = 0; j < 4; ++j)     // K: 32 rows x 256B
            async_copy16(gK + (size_t)s0 * 512 + j * 4096, &shK[buf][lK + j * 1024]);
        #pragma unroll
        for (int j = 0; j < 4; ++j)     // V^T: 128 rows x 64B
            async_copy16(gV + (size_t)s0 + j * 32 * 2048, &shVT[buf][lV + j * 1024]);
    };

    // stage Q tile once + first K/V tile
    #pragma unroll
    for (int j = 0; j < 8; ++j)
        async_copy16(gQ + (size_t)j * 8 * 2048, shQ + lK + j * 1024);
    issue_kv(0, 0);
    wait_async();
    __syncthreads();

    v8f oacc[8] = {};
    float m_old = -1e30f, l_old = 0.0f;

    const int sEnd = m0 + 64;
    for (int s0 = 0; s0 < sEnd; s0 += 32) {
        const int cur = (s0 >> 5) & 1;
        if (s0 + 32 < sEnd) issue_kv(s0 + 32, cur ^ 1);   // overlap with compute

        const unsigned short* pK = shK[cur];
        const unsigned short* pV = shVT[cur];

        // S = Q K^T : 16x32 per wave, d=128 in 4 chunks
        v8f sacc[2] = {};
        #pragma unroll
        for (int kc = 0; kc < 4; ++kc) {
            v16bf aq = load_a_frag(shQ, w * 16 + lh, 128, kc * 32 + (hi ? 8 : 0));
            #pragma unroll
            for (int st = 0; st < 2; ++st) {
                v16bf bk = load_b_frag(pK, st * 16 + lh, 128, kc * 32 + (hi ? 8 : 0));
                sacc[st] = wmma_bf16(aq, bk, sacc[st]);
            }
        }
        #pragma unroll
        for (int st = 0; st < 2; ++st)
            #pragma unroll
            for (int e = 0; e < 8; ++e)
                shS[(w * 16 + (hi ? 8 : 0) + e) * 32 + st * 16 + lh] = sacc[st][e];
        __syncthreads();

        // online softmax: lanes 0..15 own one row each
        if (lane < 16) {
            const int rg = m0 + w * 16 + lh;
            float mnew = m_old;
            float vals[32];
            #pragma unroll
            for (int j = 0; j < 32; ++j) {
                float sv = shS[(w * 16 + lh) * 32 + j];
                sv = (s0 + j <= rg) ? sv : -1e30f;        // causal mask
                vals[j] = sv;
                mnew = fmaxf(mnew, sv);
            }
            const float corr = __expf(m_old - mnew);
            float ls = 0.0f;
            #pragma unroll
            for (int j = 0; j < 32; ++j) {
                float pj = __expf(vals[j] - mnew);
                ls += pj;
                shP[(w * 16 + lh) * 32 + j] = f2bf(pj);
            }
            l_old = l_old * corr + ls;
            m_old = mnew;
            shCorr[w * 16 + lh] = corr;
            shL[w * 16 + lh]    = l_old;
        }
        __syncthreads();

        // O = corr*O + P V
        float cvals[8];
        #pragma unroll
        for (int e = 0; e < 8; ++e) cvals[e] = shCorr[w * 16 + (hi ? 8 : 0) + e];
        v16bf ap = load_a_frag(shP, w * 16 + lh, 32, hi ? 8 : 0);
        #pragma unroll
        for (int dt = 0; dt < 8; ++dt) {
            v8f o = oacc[dt];
            #pragma unroll
            for (int e = 0; e < 8; ++e) o[e] *= cvals[e];
            v16bf bv = load_b_frag(pV, dt * 16 + lh, 32, hi ? 8 : 0);
            oacc[dt] = wmma_bf16(ap, bv, o);
        }

        wait_async();
        __syncthreads();
    }

    // final 1/l scaling, write bf16 y[b, t, h*128+d]
    float lv[8];
    #pragma unroll
    for (int e = 0; e < 8; ++e) lv[e] = 1.0f / shL[w * 16 + (hi ? 8 : 0) + e];
    #pragma unroll
    for (int dt = 0; dt < 8; ++dt)
        #pragma unroll
        for (int e = 0; e < 8; ++e)
            Yb[(size_t)(b * 2048 + m0 + w * 16 + (hi ? 8 : 0) + e) * 2048 + h * 128 + dt * 16 + lh] =
                f2bf(oacc[dt][e] * lv[e]);
}

// =====================================================================
extern "C" void kernel_launch(void* const* d_in, const int* in_sizes, int n_in,
                              void* d_out, int out_size, void* d_ws, size_t ws_size,
                              hipStream_t stream)
{
    (void)in_sizes; (void)n_in; (void)out_size; (void)ws_size;
    const float* x    = (const float*)d_in[0];
    const float* cosb = (const float*)d_in[1];
    const float* sinb = (const float*)d_in[2];
    const float* Wq   = (const float*)d_in[3];
    const float* Wk   = (const float*)d_in[4];
    const float* Wv   = (const float*)d_in[5];
    const float* Wo   = (const float*)d_in[6];
    float* out = (float*)d_out;

    // workspace layout
    float* Qf = (float*)d_ws;                                  // 4096x2048 f32
    float* Kf = Qf + (size_t)4096 * 2048;                      // 4096x512  f32
    float* Vf = Kf + (size_t)4096 * 512;                       // 4096x512  f32
    unsigned short* xb  = (unsigned short*)(Vf + (size_t)4096 * 512);
    unsigned short* Wqt = xb  + (size_t)4096 * 2048;           // [2048][2048]
    unsigned short* Wkt = Wqt + (size_t)2048 * 2048;           // [512][2048]
    unsigned short* Wvt = Wkt + (size_t)512 * 2048;            // [512][2048]
    unsigned short* Wot = Wvt + (size_t)512 * 2048;            // [2048][2048]
    unsigned short* Qb  = Wot + (size_t)2048 * 2048;           // 4096x2048
    unsigned short* Kb  = Qb  + (size_t)4096 * 2048;           // 4096x512
    unsigned short* Vt  = Kb  + (size_t)4096 * 512;            // (B,4,128,T)
    unsigned short* Yb  = Vt  + (size_t)4096 * 512;            // 4096x2048

    // prep: convert / transpose to bf16
    cvt_bf16_kernel<<<(4096 * 2048) / 256, 256, 0, stream>>>(x, xb, 4096 * 2048);
    transpose_cvt_kernel<<<dim3(64, 64), 256, 0, stream>>>(Wq, Wqt, 2048, 2048);
    transpose_cvt_kernel<<<dim3(16, 64), 256, 0, stream>>>(Wk, Wkt, 2048, 512);
    transpose_cvt_kernel<<<dim3(16, 64), 256, 0, stream>>>(Wv, Wvt, 2048, 512);
    transpose_cvt_kernel<<<dim3(64, 64), 256, 0, stream>>>(Wo, Wot, 2048, 2048);

    // projections (bf16 WMMA, f32 accum)
    gemm_bf16_kernel<2048, 2048><<<dim3(16, 32), 256, 0, stream>>>(xb, Wqt, Qf);
    gemm_bf16_kernel< 512, 2048><<<dim3( 4, 32), 256, 0, stream>>>(xb, Wkt, Kf);
    gemm_bf16_kernel< 512, 2048><<<dim3( 4, 32), 256, 0, stream>>>(xb, Wvt, Vf);

    // RoPE + RMS-norm -> bf16 (1/sqrt(128) folded into Q)
    rope_rms_kernel<<<2 * 2048 * 16, 64, 0, stream>>>(Qf, Qb, cosb, sinb, 16, 0.08838834764831845f);
    rope_rms_kernel<<<2 * 2048 * 4,  64, 0, stream>>>(Kf, Kb, cosb, sinb, 4, 1.0f);
    vtrans_kernel<<<dim3(64, 4, 8), 256, 0, stream>>>(Vf, Vt);

    // causal attention
    attn_kernel<<<dim3(2048 / 64, 16, 2), 128, 0, stream>>>(Qb, Kb, Vt, Yb);

    // output projection
    gemm_bf16_kernel<2048, 2048><<<dim3(16, 32), 256, 0, stream>>>(Yb, Wot, out);
}
